// AttentionBlock_62715112456866
// MI455X (gfx1250) — matrix operand order repaired and checked by reading, not compiled, not run on hardware
//
#include <hip/hip_runtime.h>
#include <math.h>

typedef _Float16 half_t;
typedef __attribute__((ext_vector_type(16))) _Float16 v16h;
typedef __attribute__((ext_vector_type(8)))  _Float16 v8h;
typedef __attribute__((ext_vector_type(8)))  float    v8f;

#define HEADS     8
#define DIM_HEAD  64
#define B_SZ      4
#define C_DIM     512
#define L_SZ      2048
#define HID       (HEADS * DIM_HEAD)   // 512
#define QKV_ROWS  (3 * HID)            // 1536
#define NGROUPS   32
#define GN_EPS    1e-5f
#define LDSB      48                   // padded LDS row stride (halves)

#define WMMA_F16(a, b, c) \
  __builtin_amdgcn_wmma_f32_16x16x32_f16(false, (a), false, (b), (short)0, (c), false, false)

// ---------------------------------------------------------------------------
// WMMA fragment loaders — every load is a 16-byte (b128) access.
// ---------------------------------------------------------------------------
__device__ __forceinline__ v16h frag_cat(v8h lo, v8h hi) {
  return __builtin_shufflevector(lo, hi, 0, 1, 2, 3, 4, 5, 6, 7,
                                 8, 9, 10, 11, 12, 13, 14, 15);
}

// A: 16x32 f16 (M x K), row-major source, leading dim lda (halves).
__device__ __forceinline__ v16h load_frag_a(const half_t* __restrict__ base, int lda) {
  int lane  = threadIdx.x & 31;
  int row   = lane & 15;
  int kbase = (lane < 16) ? 0 : 8;
  const half_t* p = base + (size_t)row * lda + kbase;
  v8h lo = *(const v8h*)(p);
  v8h hi = *(const v8h*)(p + 16);
  return frag_cat(lo, hi);
}

__device__ __forceinline__ v16h load_frag_a_lds(const half_t* base, int lda) {
  int lane  = threadIdx.x & 31;
  int row   = lane & 15;
  int kbase = (lane < 16) ? 0 : 8;
  const half_t* p = base + (size_t)row * lda + kbase;
  v8h lo = *(const v8h*)(p);
  v8h hi = *(const v8h*)(p + 16);
  return frag_cat(lo, hi);
}

// B: 32x16 f16 (K x N), source stored N-major: element(k, n) = base[n*ld + k].
__device__ __forceinline__ v16h load_frag_bt(const half_t* __restrict__ base, int ld) {
  int lane  = threadIdx.x & 31;
  int col   = lane & 15;
  int kbase = (lane < 16) ? 0 : 8;
  const half_t* p = base + (size_t)col * ld + kbase;
  v8h lo = *(const v8h*)(p);
  v8h hi = *(const v8h*)(p + 16);
  return frag_cat(lo, hi);
}

__device__ __forceinline__ v16h load_frag_bt_lds(const half_t* base, int ld) {
  int lane  = threadIdx.x & 31;
  int col   = lane & 15;
  int kbase = (lane < 16) ? 0 : 8;
  const half_t* p = base + (size_t)col * ld + kbase;
  v8h lo = *(const v8h*)(p);
  v8h hi = *(const v8h*)(p + 16);
  return frag_cat(lo, hi);
}

// ---------------------------------------------------------------------------
// 1) GroupNorm -> f16, written TRANSPOSED as nx^T [B, L, C]
// ---------------------------------------------------------------------------
__global__ __launch_bounds__(256) void gn_kernel(const float* __restrict__ x,
                                                 const float* __restrict__ gw,
                                                 const float* __restrict__ gb,
                                                 half_t* __restrict__ nxt) {
  const int CG = C_DIM / NGROUPS;  // 16
  int b = blockIdx.x / NGROUPS;
  int g = blockIdx.x % NGROUPS;
  const float* xp = x + ((size_t)b * C_DIM + g * CG) * L_SZ;
  int tid = threadIdx.x;
  const int N = CG * L_SZ;  // 32768
  float s = 0.f, s2 = 0.f;
  for (int i = tid; i < N; i += 256) {
    float v = xp[i];
    s += v;
    s2 += v * v;
  }
  __shared__ float red0[256], red1[256];
  red0[tid] = s;
  red1[tid] = s2;
  __syncthreads();
  for (int st = 128; st > 0; st >>= 1) {
    if (tid < st) {
      red0[tid] += red0[tid + st];
      red1[tid] += red1[tid + st];
    }
    __syncthreads();
  }
  float mu  = red0[0] * (1.0f / (float)N);
  float var = red1[0] * (1.0f / (float)N) - mu * mu;
  float rs  = rsqrtf(var + GN_EPS);
  half_t* np = nxt + (size_t)b * L_SZ * C_DIM + g * CG;
  for (int i = tid; i < N; i += 256) {
    int cl = i >> 11;
    int l  = i & (L_SZ - 1);
    int c  = g * CG + cl;
    float v = (xp[i] - mu) * rs * gw[c] + gb[c];
    np[(size_t)l * C_DIM + cl] = (half_t)v;
  }
}

// ---------------------------------------------------------------------------
// 2) Weight f32 -> f16 conversion
// ---------------------------------------------------------------------------
__global__ __launch_bounds__(256) void cvt_kernel(const float* __restrict__ wq,
                                                  const float* __restrict__ wo,
                                                  half_t* __restrict__ wqh,
                                                  half_t* __restrict__ woh) {
  int i = blockIdx.x * 256 + threadIdx.x;
  if (i < QKV_ROWS * C_DIM) wqh[i] = (half_t)wq[i];
  if (i < C_DIM * HID) woh[i] = (half_t)wo[i];
}

// ---------------------------------------------------------------------------
// 3) QKV GEMM: block computes 128M x 64N; B staged per-block in double-
//    buffered LDS; A ping-pongs in registers.
// ---------------------------------------------------------------------------
__global__ __launch_bounds__(256) void qkv_kernel(const half_t* __restrict__ W,
                                                  const half_t* __restrict__ NXT,
                                                  half_t* __restrict__ Q,
                                                  half_t* __restrict__ K,
                                                  half_t* __restrict__ Vt) {
  __shared__ __align__(16) half_t bs[2][64 * LDSB];  // 12 KB
  const int MB  = (QKV_ROWS / 16) / 8;  // 12
  const int NT4 = L_SZ / 64;            // 32

  int blk  = blockIdx.x;
  int b    = blk / (MB * NT4);
  int rem  = blk % (MB * NT4);
  int mblk = rem / NT4;
  int nt4  = rem % NT4;
  int w    = threadIdx.x >> 5;
  int mt   = mblk * 8 + w;

  int tid  = threadIdx.x;
  int srow = tid >> 2;
  int sseg = (tid & 3) * 8;

  const half_t* Ab   = W + (size_t)mt * 16 * C_DIM;
  const half_t* Bsrc = NXT + (size_t)b * L_SZ * C_DIM + (size_t)nt4 * 64 * C_DIM;

  {
    v8h st = *(const v8h*)(Bsrc + (size_t)srow * C_DIM + sseg);
    *(v8h*)&bs[0][srow * LDSB + sseg] = st;
  }
  v16h a_cur = load_frag_a(Ab, C_DIM);
  __syncthreads();

  v8f acc[4] = {{}, {}, {}, {}};
  int buf = 0;
#pragma unroll 2
  for (int c = 0; c < C_DIM; c += 32) {
    int cn = (c + 32) & (C_DIM - 1);
    v8h  st    = *(const v8h*)(Bsrc + (size_t)srow * C_DIM + cn + sseg);
    v16h a_nxt = load_frag_a(Ab + cn, C_DIM);
    const half_t* bp = &bs[buf][0];
    v16h b0 = load_frag_bt_lds(bp, LDSB);
    v16h b1 = load_frag_bt_lds(bp + 16 * LDSB, LDSB);
    v16h b2 = load_frag_bt_lds(bp + 32 * LDSB, LDSB);
    v16h b3 = load_frag_bt_lds(bp + 48 * LDSB, LDSB);
    acc[0] = WMMA_F16(a_cur, b0, acc[0]);
    acc[1] = WMMA_F16(a_cur, b1, acc[1]);
    acc[2] = WMMA_F16(a_cur, b2, acc[2]);
    acc[3] = WMMA_F16(a_cur, b3, acc[3]);
    buf ^= 1;
    *(v8h*)&bs[buf][srow * LDSB + sseg] = st;
    a_cur = a_nxt;
    __syncthreads();
  }

  int lane  = threadIdx.x & 31;
  int col   = lane & 15;
  int mbase = (lane < 16) ? 0 : 8;
  int o0    = mt * 16 + mbase;
  const float scale = 0.125f;  // DIM_HEAD^-0.5

  if (mt < 32) {
    int h = o0 >> 6, d0 = o0 & 63;
#pragma unroll
    for (int i = 0; i < 4; ++i) {
      int l = nt4 * 64 + i * 16 + col;
      v8h pk;
#pragma unroll
      for (int r = 0; r < 8; ++r) pk[r] = (half_t)(acc[i][r] * scale);
      *(v8h*)&Q[(((size_t)b * HEADS + h) * L_SZ + l) * DIM_HEAD + d0] = pk;
    }
  } else if (mt < 64) {
    int oo = o0 - HID;
    int h = oo >> 6, d0 = oo & 63;
#pragma unroll
    for (int i = 0; i < 4; ++i) {
      int l = nt4 * 64 + i * 16 + col;
      v8h pk;
#pragma unroll
      for (int r = 0; r < 8; ++r) pk[r] = (half_t)acc[i][r];
      *(v8h*)&K[(((size_t)b * HEADS + h) * L_SZ + l) * DIM_HEAD + d0] = pk;
    }
  } else {
    int oo = o0 - 2 * HID;
    int h = oo >> 6, d0 = oo & 63;
    half_t* vb = Vt + (((size_t)b * HEADS + h) * DIM_HEAD + d0) * L_SZ;
#pragma unroll
    for (int i = 0; i < 4; ++i) {
      int l = nt4 * 64 + i * 16 + col;
#pragma unroll
      for (int r = 0; r < 8; ++r) vb[(size_t)r * L_SZ + l] = (half_t)acc[i][r];
    }
  }
}

// ---------------------------------------------------------------------------
// 4) Flash attention, 64-column j-tiles: softmax bookkeeping (max-shuffles,
//    alpha rescale) amortized over 2x the columns; row sums via P x ones
//    WMMAs. 18 WMMAs per 64 columns.
// ---------------------------------------------------------------------------
__global__ __launch_bounds__(256) void attn_kernel(const half_t* __restrict__ Q,
                                                   const half_t* __restrict__ K,
                                                   const half_t* __restrict__ Vt,
                                                   half_t* __restrict__ AOT) {
  __shared__ __align__(16) half_t plds[8][16 * 64];  // per-wave 16x64 P tile
  int w    = threadIdx.x >> 5;
  int lane = threadIdx.x & 31;

  int bh   = blockIdx.x / (L_SZ / 128);
  int qt   = blockIdx.x % (L_SZ / 128);
  int qrow = qt * 128 + w * 16;

  const half_t* Qp = Q + (size_t)bh * L_SZ * DIM_HEAD;
  const half_t* Kp = K + (size_t)bh * L_SZ * DIM_HEAD;
  const half_t* Vp = Vt + (size_t)bh * DIM_HEAD * L_SZ;

  v16h aq0 = load_frag_a(Qp + (size_t)qrow * DIM_HEAD, DIM_HEAD);
  v16h aq1 = load_frag_a(Qp + (size_t)qrow * DIM_HEAD + 32, DIM_HEAD);

  v16h ones;
#pragma unroll
  for (int i = 0; i < 16; ++i) ones[i] = (half_t)1.0f;

  v8f o0 = {}, o1 = {}, o2 = {}, o3 = {};
  v8f osum = {};  // row sums via P x ones (rescaled with same alpha)
  float mrow[8];
#pragma unroll
  for (int r = 0; r < 8; ++r) mrow[r] = -INFINITY;

  int col   = lane & 15;
  int mbase = (lane < 16) ? 0 : 8;

#pragma unroll 1
  for (int j = 0; j < L_SZ; j += 64) {
    if (j + 64 < L_SZ) {
      __builtin_prefetch(Kp + (size_t)(j + 64 + 2 * lane) * DIM_HEAD, 0, 3);
      __builtin_prefetch(Kp + (size_t)(j + 64 + 2 * lane) * DIM_HEAD + 64, 0, 3);
      __builtin_prefetch(Vp + (size_t)lane * L_SZ + j + 64, 0, 3);
      __builtin_prefetch(Vp + (size_t)(32 + lane) * L_SZ + j + 64, 0, 3);
    }

    // ---- S = Q . K^T : four 16x16 column tiles ----
    v8f s0 = {}, s1 = {}, s2 = {}, s3 = {};
    {
      v16h bk = load_frag_bt(Kp + (size_t)j * DIM_HEAD, DIM_HEAD);
      s0 = WMMA_F16(aq0, bk, s0);
      bk = load_frag_bt(Kp + (size_t)j * DIM_HEAD + 32, DIM_HEAD);
      s0 = WMMA_F16(aq1, bk, s0);
      bk = load_frag_bt(Kp + (size_t)(j + 16) * DIM_HEAD, DIM_HEAD);
      s1 = WMMA_F16(aq0, bk, s1);
      bk = load_frag_bt(Kp + (size_t)(j + 16) * DIM_HEAD + 32, DIM_HEAD);
      s1 = WMMA_F16(aq1, bk, s1);
      bk = load_frag_bt(Kp + (size_t)(j + 32) * DIM_HEAD, DIM_HEAD);
      s2 = WMMA_F16(aq0, bk, s2);
      bk = load_frag_bt(Kp + (size_t)(j + 32) * DIM_HEAD + 32, DIM_HEAD);
      s2 = WMMA_F16(aq1, bk, s2);
      bk = load_frag_bt(Kp + (size_t)(j + 48) * DIM_HEAD, DIM_HEAD);
      s3 = WMMA_F16(aq0, bk, s3);
      bk = load_frag_bt(Kp + (size_t)(j + 48) * DIM_HEAD + 32, DIM_HEAD);
      s3 = WMMA_F16(aq1, bk, s3);
    }

    // ---- online softmax: one max/alpha per row per 64 columns ----
#pragma unroll
    for (int r = 0; r < 8; ++r) {
      float t = fmaxf(fmaxf(s0[r], s1[r]), fmaxf(s2[r], s3[r]));
      t = fmaxf(t, __shfl_xor(t, 1, 32));
      t = fmaxf(t, __shfl_xor(t, 2, 32));
      t = fmaxf(t, __shfl_xor(t, 4, 32));
      t = fmaxf(t, __shfl_xor(t, 8, 32));
      float mn    = fmaxf(mrow[r], t);
      float alpha = __expf(mrow[r] - mn);
      mrow[r]     = mn;
      float p0 = __expf(s0[r] - mn);
      float p1 = __expf(s1[r] - mn);
      float p2 = __expf(s2[r] - mn);
      float p3 = __expf(s3[r] - mn);
      o0[r] *= alpha;
      o1[r] *= alpha;
      o2[r] *= alpha;
      o3[r] *= alpha;
      osum[r] *= alpha;
      int row = mbase + r;
      plds[w][row * 64 + col]      = (half_t)p0;
      plds[w][row * 64 + col + 16] = (half_t)p1;
      plds[w][row * 64 + col + 32] = (half_t)p2;
      plds[w][row * 64 + col + 48] = (half_t)p3;
    }

    // ---- O += P . V ; osum += P . 1 (two K=32 halves) ----
    v16h ap0 = load_frag_a_lds(&plds[w][0], 64);   // j-cols 0..31
    v16h ap1 = load_frag_a_lds(&plds[w][32], 64);  // j-cols 32..63
    osum = WMMA_F16(ap0, ones, osum);
    osum = WMMA_F16(ap1, ones, osum);
    v16h bv = load_frag_bt(Vp + j, L_SZ);
    o0 = WMMA_F16(ap0, bv, o0);
    bv = load_frag_bt(Vp + j + 32, L_SZ);
    o0 = WMMA_F16(ap1, bv, o0);
    bv = load_frag_bt(Vp + 16 * L_SZ + j, L_SZ);
    o1 = WMMA_F16(ap0, bv, o1);
    bv = load_frag_bt(Vp + 16 * L_SZ + j + 32, L_SZ);
    o1 = WMMA_F16(ap1, bv, o1);
    bv = load_frag_bt(Vp + 32 * L_SZ + j, L_SZ);
    o2 = WMMA_F16(ap0, bv, o2);
    bv = load_frag_bt(Vp + 32 * L_SZ + j + 32, L_SZ);
    o2 = WMMA_F16(ap1, bv, o2);
    bv = load_frag_bt(Vp + 48 * L_SZ + j, L_SZ);
    o3 = WMMA_F16(ap0, bv, o3);
    bv = load_frag_bt(Vp + 48 * L_SZ + j + 32, L_SZ);
    o3 = WMMA_F16(ap1, bv, o3);
  }

  // ---- epilogue: normalize by osum, store transposed ao^T[B, L, C] ----
  int b = bh / HEADS, h = bh % HEADS;
#pragma unroll
  for (int r = 0; r < 8; ++r) {
    float inv = 1.0f / osum[r];
    int i = qrow + mbase + r;
    half_t* ob = AOT + ((size_t)b * L_SZ + i) * C_DIM + h * DIM_HEAD;
    ob[col]      = (half_t)(o0[r] * inv);
    ob[col + 16] = (half_t)(o1[r] * inv);
    ob[col + 32] = (half_t)(o2[r] * inv);
    ob[col + 48] = (half_t)(o3[r] * inv);
  }
}

// ---------------------------------------------------------------------------
// 5) Projection GEMM (LDS-staged) + bias + residual, f32 out.
// ---------------------------------------------------------------------------
__global__ __launch_bounds__(256) void proj_kernel(const half_t* __restrict__ W,
                                                   const half_t* __restrict__ AOT,
                                                   const float* __restrict__ bias,
                                                   const float* __restrict__ x,
                                                   float* __restrict__ out) {
  __shared__ __align__(16) half_t bs[2][64 * LDSB];
  const int MB  = (C_DIM / 16) / 8;  // 4
  const int NT4 = L_SZ / 64;         // 32

  int blk  = blockIdx.x;
  int b    = blk / (MB * NT4);
  int rem  = blk % (MB * NT4);
  int mblk = rem / NT4;
  int nt4  = rem % NT4;
  int w    = threadIdx.x >> 5;
  int mt   = mblk * 8 + w;

  int tid  = threadIdx.x;
  int srow = tid >> 2;
  int sseg = (tid & 3) * 8;

  const half_t* Ab   = W + (size_t)mt * 16 * HID;
  const half_t* Bsrc = AOT + (size_t)b * L_SZ * C_DIM + (size_t)nt4 * 64 * C_DIM;

  {
    v8h st = *(const v8h*)(Bsrc + (size_t)srow * C_DIM + sseg);
    *(v8h*)&bs[0][srow * LDSB + sseg] = st;
  }
  v16h a_cur = load_frag_a(Ab, HID);
  __syncthreads();

  v8f acc[4] = {{}, {}, {}, {}};
  int buf = 0;
#pragma unroll 2
  for (int c = 0; c < HID; c += 32) {
    int cn = (c + 32) & (HID - 1);
    v8h  st    = *(const v8h*)(Bsrc + (size_t)srow * C_DIM + cn + sseg);
    v16h a_nxt = load_frag_a(Ab + cn, HID);
    const half_t* bp = &bs[buf][0];
    v16h b0 = load_frag_bt_lds(bp, LDSB);
    v16h b1 = load_frag_bt_lds(bp + 16 * LDSB, LDSB);
    v16h b2 = load_frag_bt_lds(bp + 32 * LDSB, LDSB);
    v16h b3 = load_frag_bt_lds(bp + 48 * LDSB, LDSB);
    acc[0] = WMMA_F16(a_cur, b0, acc[0]);
    acc[1] = WMMA_F16(a_cur, b1, acc[1]);
    acc[2] = WMMA_F16(a_cur, b2, acc[2]);
    acc[3] = WMMA_F16(a_cur, b3, acc[3]);
    buf ^= 1;
    *(v8h*)&bs[buf][srow * LDSB + sseg] = st;
    a_cur = a_nxt;
    __syncthreads();
  }

  int lane  = threadIdx.x & 31;
  int col   = lane & 15;
  int mbase = (lane < 16) ? 0 : 8;
#pragma unroll
  for (int i = 0; i < 4; ++i) {
    int l = nt4 * 64 + i * 16 + col;
#pragma unroll
    for (int r = 0; r < 8; ++r) {
      int m = mt * 16 + mbase + r;
      size_t idx = ((size_t)b * C_DIM + m) * L_SZ + l;
      out[idx] = acc[i][r] + bias[m] + x[idx];
    }
  }
}

// ---------------------------------------------------------------------------
// Launch
// ---------------------------------------------------------------------------
extern "C" void kernel_launch(void* const* d_in, const int* in_sizes, int n_in,
                              void* d_out, int out_size, void* d_ws, size_t ws_size,
                              hipStream_t stream) {
  const float* x     = (const float*)d_in[0];
  const float* gn_w  = (const float*)d_in[1];
  const float* gn_b  = (const float*)d_in[2];
  const float* w_qkv = (const float*)d_in[3];
  const float* w_out = (const float*)d_in[4];
  const float* b_out = (const float*)d_in[5];
  float* out = (float*)d_out;

  char* ws = (char*)d_ws;
  const size_t act_bytes = (size_t)B_SZ * C_DIM * L_SZ * sizeof(half_t);  // 8 MB each
  half_t* nxt   = (half_t*)(ws);                    // nx^T  [B, L, C]
  half_t* qbuf  = (half_t*)(ws + act_bytes);        // Q     [B, H, L, D]
  half_t* kbuf  = (half_t*)(ws + 2 * act_bytes);    // K     [B, H, L, D]
  half_t* vtbuf = (half_t*)(ws + 3 * act_bytes);    // V^T   [B, H, D, L]
  half_t* aot   = (half_t*)(ws + 4 * act_bytes);    // ao^T  [B, L, C]
  half_t* wqh   = (half_t*)(ws + 5 * act_bytes);
  half_t* woh   = (half_t*)(ws + 5 * act_bytes + (size_t)QKV_ROWS * C_DIM * sizeof(half_t));

  gn_kernel<<<B_SZ * NGROUPS, 256, 0, stream>>>(x, gn_w, gn_b, nxt);
  cvt_kernel<<<(QKV_ROWS * C_DIM + 255) / 256, 256, 0, stream>>>(w_qkv, w_out, wqh, woh);

  {
    int blocks = B_SZ * ((QKV_ROWS / 16) / 8) * (L_SZ / 64);
    qkv_kernel<<<blocks, 256, 0, stream>>>(wqh, nxt, qbuf, kbuf, vtbuf);
  }

  attn_kernel<<<B_SZ * HEADS * (L_SZ / 128), 256, 0, stream>>>(qbuf, kbuf, vtbuf, aot);

  {
    int blocks = B_SZ * ((C_DIM / 16) / 8) * (L_SZ / 64);
    proj_kernel<<<blocks, 256, 0, stream>>>(woh, aot, b_out, x, out);
  }
}
